// CausalSelectiveSelfAttention_19859928776994
// MI455X (gfx1250) — matrix operand order repaired and checked by reading, not compiled, not run on hardware
//
#include <hip/hip_runtime.h>

typedef __attribute__((ext_vector_type(16))) _Float16 v16h;
typedef __attribute__((ext_vector_type(8)))  _Float16 v8h;
typedef __attribute__((ext_vector_type(8)))  float    v8f;

#define WMMA_F16(a, b, c) \
    __builtin_amdgcn_wmma_f32_16x16x32_f16(false, (a), false, (b), (short)0, (c), false, false)

// Async global->LDS copy, 16B per lane, immediate offset applies to both sides.
#define ASYNC_B128(ldsoff, gaddr, imm)                                      \
    asm volatile("global_load_async_to_lds_b128 %0, %1, off offset:" #imm   \
                 :: "v"(ldsoff), "v"(gaddr) : "memory")

constexpr int Bc = 2, Tc = 2048, Cc = 768, Hc = 12, Dc = 64;
constexpr int HD = Hc * Dc;          // 768
constexpr int NQKV = 3 * HD;         // 2304
constexpr float ATTN_MULT = 0.125f;
constexpr int WPB = 8;               // waves per block for GEMMs
constexpr int AWPB = 4;              // waves per block for attention (bigger LDS/wave)

__device__ inline v8f zero8() {
    v8f z;
#pragma unroll
    for (int i = 0; i < 8; ++i) z[i] = 0.0f;
    return z;
}

__device__ inline float redmax16(float v) {
    v = fmaxf(v, __shfl_xor(v, 1));
    v = fmaxf(v, __shfl_xor(v, 2));
    v = fmaxf(v, __shfl_xor(v, 4));
    v = fmaxf(v, __shfl_xor(v, 8));
    return v;
}
__device__ inline float redsum16(float v) {
    v += __shfl_xor(v, 1);
    v += __shfl_xor(v, 2);
    v += __shfl_xor(v, 4);
    v += __shfl_xor(v, 8);
    return v;
}

// A-fragment (16x32 f16, row-major source).
// lane<16: row=lane, K = [k0+hl*16 .. +15] contiguous  (hl = lane>>4)
__device__ inline v16h load_a_frag(const _Float16* __restrict__ rowp, int k0, int hl) {
    const _Float16* p = rowp + k0 + hl * 16;
    v8h lo = *(const v8h*)p;
    v8h hi = *(const v8h*)(p + 8);
    v16h r;
#pragma unroll
    for (int e = 0; e < 8; ++e) { r[e] = lo[e]; r[e + 8] = hi[e]; }
    return r;
}

// ---------------------------------------------------------------- converts
__global__ void cvt_f32_f16_kernel(const float* __restrict__ in,
                                   _Float16* __restrict__ out, int n) {
    for (int i = blockIdx.x * blockDim.x + threadIdx.x; i < n;
         i += gridDim.x * blockDim.x)
        out[i] = (_Float16)in[i];
}

// ---------------------------------------------------------------- QKV GEMM
// qkv = x @ w_attn^T + b_attn ; scatter into head-major q/k/v [B,H,T,D] f16
// Explicit ping-pong double buffering (no inter-buffer register moves).
__global__ __launch_bounds__(256) void qkv_gemm_kernel(
    const _Float16* __restrict__ xh, const _Float16* __restrict__ wh,
    const float* __restrict__ bias, _Float16* __restrict__ qh,
    _Float16* __restrict__ kh, _Float16* __restrict__ vh) {
    const int lane = threadIdx.x & 31, wave = threadIdx.x >> 5;
    const int gw = blockIdx.x * WPB + wave;
    const int MT = (Bc * Tc) / 16;                       // 256
    const int mt = gw % MT;
    const int ng = gw / MT;                              // 0..35 (64-col groups)
    const int hl = lane >> 4, lr = lane & 15;
    const _Float16* arow = xh + (size_t)(mt * 16 + lr) * Cc;
    const _Float16* brow[4];
#pragma unroll
    for (int c = 0; c < 4; ++c)
        brow[c] = wh + (size_t)(ng * 64 + c * 16 + lr) * Cc + hl * 16;

    v8f acc[4];
#pragma unroll
    for (int c = 0; c < 4; ++c) acc[c] = zero8();

    v16h a0 = load_a_frag(arow, 0, hl);
    v16h b0[4];
#pragma unroll
    for (int c = 0; c < 4; ++c) b0[c] = *(const v16h*)(brow[c]);

    for (int k = 0; k < Cc; k += 64) {                   // 12 iterations (768/64)
        int k1 = k + 32;                                 // always < Cc
        v16h a1 = load_a_frag(arow, k1, hl);
        v16h b1[4];
#pragma unroll
        for (int c = 0; c < 4; ++c) b1[c] = *(const v16h*)(brow[c] + k1);
#pragma unroll
        for (int c = 0; c < 4; ++c) acc[c] = WMMA_F16(a0, b0[c], acc[c]);

        int k2 = k + 64; if (k2 >= Cc) k2 = Cc - 32;     // harmless tail reload
        a0 = load_a_frag(arow, k2, hl);
#pragma unroll
        for (int c = 0; c < 4; ++c) b0[c] = *(const v16h*)(brow[c] + k2);
#pragma unroll
        for (int c = 0; c < 4; ++c) acc[c] = WMMA_F16(a1, b1[c], acc[c]);
    }
#pragma unroll
    for (int c = 0; c < 4; ++c) {
        int n = ng * 64 + c * 16 + lr;
        int part = n / HD, rem = n % HD;
        int h = rem / Dc, d = rem % Dc;
        _Float16* dst = (part == 0) ? qh : ((part == 1) ? kh : vh);
        float bv = bias[n];
#pragma unroll
        for (int r = 0; r < 8; ++r) {
            int m = mt * 16 + r + hl * 8;          // C layout: row = r + 8*(lane>=16)
            int bb = m / Tc, t = m % Tc;
            dst[(((size_t)bb * Hc + h) * Tc + t) * Dc + d] =
                (_Float16)(acc[c][r] + bv);
        }
    }
}

// ---------------------------------------------------------------- Fmask scan
// Fmask[b,i,j] = sum_{i'<i} relu(0.125 * q0[i'] . k0[j]),  valid iff 1<=j<i'
__global__ __launch_bounds__(256) void fmask_kernel(
    const _Float16* __restrict__ qh, const _Float16* __restrict__ kh,
    float* __restrict__ fm) {
    int idx = blockIdx.x * 256 + threadIdx.x;   // one thread per (b, key column j)
    int b = idx / Tc, j = idx % Tc;
    const _Float16* kr = kh + ((size_t)b * Hc) * Tc * Dc + (size_t)j * Dc;
    v8h kreg[8];
#pragma unroll
    for (int w = 0; w < 8; ++w) kreg[w] = *(const v8h*)(kr + w * 8);
    const _Float16* q0 = qh + ((size_t)b * Hc) * Tc * Dc;
    float* fmb = fm + (size_t)b * Tc * Tc + j;
    float run = 0.0f;
    for (int i = 0; i < Tc; ++i) {
        fmb[(size_t)i * Tc] = run;              // exclusive prefix
        if (j >= 1 && j < i) {
            const _Float16* qr = q0 + (size_t)i * Dc;
            float dot = 0.0f;
#pragma unroll
            for (int w = 0; w < 8; ++w) {
                v8h qv = *(const v8h*)(qr + w * 8);
#pragma unroll
                for (int e = 0; e < 8; ++e) dot += (float)qv[e] * (float)kreg[w][e];
            }
            float s = dot * ATTN_MULT;
            if (s > 0.0f) run += s;
        }
    }
}

// ---------------------------------------------------------------- attention
// One wave per 16-query tile per (b,h). Streaming keys in chunks of 32.
// V tiles double-buffered in LDS via async global->LDS: chunk i+1 streams in
// (ASYNCcnt) while chunk i computes QK^T + softmax; fence waits asynccnt<=8.
__global__ __launch_bounds__(128) void attn_kernel(
    const _Float16* __restrict__ qh, const _Float16* __restrict__ kh,
    const _Float16* __restrict__ vh, const float* __restrict__ fm,
    _Float16* __restrict__ yh) {
    __shared__ _Float16 vLds[AWPB][2][32 * 64];   // double-buffered V tiles
    __shared__ _Float16 pLds[AWPB][16 * 32];
    const int lane = threadIdx.x & 31, wave = threadIdx.x >> 5;
    const int gw = blockIdx.x * AWPB + wave;
    const int QT = Tc / 16;                               // 128
    const int qt = gw % QT, bh = gw / QT;                 // bh in [0, B*H)
    const int b = bh / Hc, h = bh % Hc;
    const int hl = lane >> 4, lr = lane & 15;
    const int qbase = qt * 16;

    const _Float16* qrow = qh + ((size_t)bh * Tc + qbase + lr) * Dc;
    v16h aq0 = load_a_frag(qrow, 0, hl);
    v16h aq1 = load_a_frag(qrow, 32, hl);

    float Mr[8], Lr[8];
    v8f O[4];
#pragma unroll
    for (int r = 0; r < 8; ++r) { Mr[r] = -1e30f; Lr[r] = 0.0f; }
#pragma unroll
    for (int c = 0; c < 4; ++c) O[c] = zero8();

    const float* fmB = fm + (size_t)b * Tc * Tc;
    const float* fmRow[8];
#pragma unroll
    for (int r = 0; r < 8; ++r)
        fmRow[r] = fmB + (size_t)(qbase + r + hl * 8) * Tc;

    _Float16* pL = &pLds[wave][0];
    const _Float16* vBase = vh + (size_t)bh * Tc * Dc;
    const unsigned vOff0 = (unsigned)(size_t)(&vLds[wave][0][0] + (size_t)lane * 64);
    const unsigned vOff1 = (unsigned)(size_t)(&vLds[wave][1][0] + (size_t)lane * 64);
    const int kend = (qt + 1) * 16;

    // issue a V-tile (32 key rows, 128B per lane) async into LDS buffer
    auto issueV = [&](int kk, unsigned ldsoff) {
        int vkey = kk + lane; if (vkey >= Tc) vkey = Tc - 1;
        unsigned long long ga = (unsigned long long)(size_t)(vBase + (size_t)vkey * Dc);
        ASYNC_B128(ldsoff, ga, 0);
        ASYNC_B128(ldsoff, ga, 16);
        ASYNC_B128(ldsoff, ga, 32);
        ASYNC_B128(ldsoff, ga, 48);
        ASYNC_B128(ldsoff, ga, 64);
        ASYNC_B128(ldsoff, ga, 80);
        ASYNC_B128(ldsoff, ga, 96);
        ASYNC_B128(ldsoff, ga, 112);
    };

    issueV(0, vOff0);                                    // prologue: chunk 0 -> buf0

    for (int kk = 0; kk < kend; kk += 32) {
        const int p = (kk >> 5) & 1;
        // --- stream next chunk's V into the other buffer (dummy reload on tail)
        int knext = kk + 32;
        issueV((knext < kend) ? knext : kk, p ? vOff0 : vOff1);

        // --- load cluster: all K B-fragments + Fmask tiles for this chunk
        const _Float16* k0row = kh + ((size_t)bh * Tc + kk + lr) * Dc;   // sub0 in-bounds
        int key1 = kk + 16 + lr;
        int key1c = (key1 >= Tc) ? Tc - 1 : key1;
        const _Float16* k1row = kh + ((size_t)bh * Tc + key1c) * Dc;
        v16h kb00 = *(const v16h*)(k0row + hl * 16);         // d 0..31
        v16h kb01 = *(const v16h*)(k0row + 32 + hl * 16);    // d 32..63
        v16h kb10 = *(const v16h*)(k1row + hl * 16);
        v16h kb11 = *(const v16h*)(k1row + 32 + hl * 16);
        float fm0[8], fm1[8];
#pragma unroll
        for (int r = 0; r < 8; ++r) {
            fm0[r] = fmRow[r][kk + lr];
            fm1[r] = fmRow[r][key1c];
        }
        // --- scores (4 WMMAs, loads above already in flight)
        v8f s0v = zero8(), s1v = zero8();
        s0v = WMMA_F16(aq0, kb00, s0v);
        s0v = WMMA_F16(aq1, kb01, s0v);
        s1v = WMMA_F16(aq0, kb10, s1v);
        s1v = WMMA_F16(aq1, kb11, s1v);

        float sv0[8], sv1[8];
#pragma unroll
        for (int r = 0; r < 8; ++r) {
            int row = qbase + r + hl * 8;
            float a0 = s0v[r] * ATTN_MULT - fm0[r];
            float a1 = s1v[r] * ATTN_MULT - fm1[r];
            if (kk + lr > row) a0 = -1e30f;                   // causal
            if (key1 > row)    a1 = -1e30f;
            sv0[r] = a0; sv1[r] = a1;
        }
        // --- online softmax per row (row r+hl*8 lives in a fixed 16-lane group)
#pragma unroll
        for (int r = 0; r < 8; ++r) {
            float mloc = redmax16(fmaxf(sv0[r], sv1[r]));
            float newM = fmaxf(Mr[r], mloc);
            float scale = __expf(Mr[r] - newM);
            float p0 = __expf(sv0[r] - newM);
            float p1 = __expf(sv1[r] - newM);
            Lr[r] = Lr[r] * scale + redsum16(p0 + p1);
            Mr[r] = newM;
#pragma unroll
            for (int c = 0; c < 4; ++c) O[c][r] *= scale;
            int rowL = r + hl * 8;
            pL[rowL * 32 + lr]      = (_Float16)p0;
            pL[rowL * 32 + 16 + lr] = (_Float16)p1;
        }
        // --- fence: current chunk's 8 asyncs done (<=8 leaves next in flight);
        //     dscnt drains the pLds stores (intra-wave RAW)
        asm volatile("s_wait_asynccnt 0x8" ::: "memory");
        asm volatile("s_wait_dscnt 0x0" ::: "memory");
        // P as A-fragment (16x32)
        v16h ap;
        {
            const _Float16* pr = pL + lr * 32 + hl * 16;
            v8h lo = *(const v8h*)pr;
            v8h hi = *(const v8h*)(pr + 8);
#pragma unroll
            for (int e = 0; e < 8; ++e) { ap[e] = lo[e]; ap[e + 8] = hi[e]; }
        }
        // O += P @ V  (V gathered as B-fragments from current LDS buffer)
        const _Float16* vL = &vLds[wave][p][0];
#pragma unroll
        for (int c = 0; c < 4; ++c) {
            const _Float16* vb = vL + (hl * 16) * 64 + c * 16 + lr;
            v16h bf;
#pragma unroll
            for (int e = 0; e < 16; ++e) bf[e] = vb[e * 64];
            O[c] = WMMA_F16(ap, bf, O[c]);
        }
    }
    // drain the tail dummy asyncs before wave exit
    asm volatile("s_wait_asynccnt 0x0" ::: "memory");
    // normalize and store y (row-major [B*T, H*D] f16 for the projection GEMM)
#pragma unroll
    for (int r = 0; r < 8; ++r) {
        float inv = (Lr[r] > 0.0f) ? 1.0f / Lr[r] : 0.0f;
        int row = qbase + r + hl * 8;
        size_t base = ((size_t)b * Tc + row) * HD + (size_t)h * Dc;
#pragma unroll
        for (int c = 0; c < 4; ++c)
            yh[base + c * 16 + lr] = (_Float16)(O[c][r] * inv);
    }
}

// ---------------------------------------------------------------- out proj
__global__ __launch_bounds__(256) void proj_gemm_kernel(
    const _Float16* __restrict__ yh, const _Float16* __restrict__ wh,
    const float* __restrict__ bias, float* __restrict__ out) {
    const int lane = threadIdx.x & 31, wave = threadIdx.x >> 5;
    const int gw = blockIdx.x * WPB + wave;
    const int MT = (Bc * Tc) / 16;                        // 256
    const int mt = gw % MT;
    const int ng = gw / MT;                               // 0..11
    const int hl = lane >> 4, lr = lane & 15;
    const _Float16* arow = yh + (size_t)(mt * 16 + lr) * HD;
    const _Float16* brow[4];
#pragma unroll
    for (int c = 0; c < 4; ++c)
        brow[c] = wh + (size_t)(ng * 64 + c * 16 + lr) * HD + hl * 16;

    v8f acc[4];
#pragma unroll
    for (int c = 0; c < 4; ++c) acc[c] = zero8();

    v16h a0 = load_a_frag(arow, 0, hl);
    v16h b0[4];
#pragma unroll
    for (int c = 0; c < 4; ++c) b0[c] = *(const v16h*)(brow[c]);

    for (int k = 0; k < HD; k += 64) {                   // 12 iterations
        int k1 = k + 32;
        v16h a1 = load_a_frag(arow, k1, hl);
        v16h b1[4];
#pragma unroll
        for (int c = 0; c < 4; ++c) b1[c] = *(const v16h*)(brow[c] + k1);
#pragma unroll
        for (int c = 0; c < 4; ++c) acc[c] = WMMA_F16(a0, b0[c], acc[c]);

        int k2 = k + 64; if (k2 >= HD) k2 = HD - 32;
        a0 = load_a_frag(arow, k2, hl);
#pragma unroll
        for (int c = 0; c < 4; ++c) b0[c] = *(const v16h*)(brow[c] + k2);
#pragma unroll
        for (int c = 0; c < 4; ++c) acc[c] = WMMA_F16(a1, b1[c], acc[c]);
    }
#pragma unroll
    for (int c = 0; c < 4; ++c) {
        int n = ng * 64 + c * 16 + lr;
        float bv = bias[n];
#pragma unroll
        for (int r = 0; r < 8; ++r) {
            int m = mt * 16 + r + hl * 8;
            out[(size_t)m * Cc + n] = acc[c][r] + bv;
        }
    }
}

// ---------------------------------------------------------------- launch
extern "C" void kernel_launch(void* const* d_in, const int* in_sizes, int n_in,
                              void* d_out, int out_size, void* d_ws, size_t ws_size,
                              hipStream_t stream) {
    const float* x      = (const float*)d_in[0];   // [B,T,C]
    const float* w_attn = (const float*)d_in[1];   // [2304,768]
    const float* b_attn = (const float*)d_in[2];   // [2304]
    const float* w_proj = (const float*)d_in[3];   // [768,768]
    const float* b_proj = (const float*)d_in[4];   // [768]
    float* out = (float*)d_out;

    char* ws = (char*)d_ws;
    size_t off = 0;
    _Float16* xh  = (_Float16*)(ws + off); off += (size_t)Bc * Tc * Cc * 2;
    _Float16* whA = (_Float16*)(ws + off); off += (size_t)NQKV * Cc * 2;
    _Float16* whP = (_Float16*)(ws + off); off += (size_t)Cc * HD * 2;
    _Float16* qh  = (_Float16*)(ws + off); off += (size_t)Bc * Hc * Tc * Dc * 2;
    _Float16* kh  = (_Float16*)(ws + off); off += (size_t)Bc * Hc * Tc * Dc * 2;
    _Float16* vh  = (_Float16*)(ws + off); off += (size_t)Bc * Hc * Tc * Dc * 2;
    float*    fmk = (float*)(ws + off);    off += (size_t)Bc * Tc * Tc * 4;
    _Float16* yh  = (_Float16*)(ws + off); off += (size_t)Bc * Tc * HD * 2;
    (void)ws_size; (void)in_sizes; (void)n_in; (void)out_size;

    int nX = Bc * Tc * Cc, nWA = NQKV * Cc, nWP = Cc * HD;
    cvt_f32_f16_kernel<<<2048, 256, 0, stream>>>(x, xh, nX);
    cvt_f32_f16_kernel<<<2048, 256, 0, stream>>>(w_attn, whA, nWA);
    cvt_f32_f16_kernel<<<1024, 256, 0, stream>>>(w_proj, whP, nWP);

    // 256 M-tiles * 36 N-groups = 9216 waves / 8 = 1152 blocks
    qkv_gemm_kernel<<<1152, 256, 0, stream>>>(xh, whA, b_attn, qh, kh, vh);

    // B*T = 4096 key columns
    fmask_kernel<<<(Bc * Tc) / 256, 256, 0, stream>>>(qh, kh, fmk);

    // B*H*(T/16) = 3072 waves / 4 = 768 blocks (128 threads each)
    attn_kernel<<<768, 128, 0, stream>>>(qh, kh, vh, fmk, yh);

    // 256 M-tiles * 12 N-groups = 3072 waves / 8 = 384 blocks
    proj_gemm_kernel<<<384, 256, 0, stream>>>(yh, whP, b_proj, out);
}
